// TriangleMultiplicativeUpdate_80152679678057
// MI455X (gfx1250) — matrix operand order repaired and checked by reading, not compiled, not run on hardware
//
#include <hip/hip_runtime.h>

// ---------------------------------------------------------------------------
// Triangle multiplicative update (outgoing), N=1, S=512, Z=E=128, MI455X.
// All contractions on v_wmma_f32_16x16x32_f16 (wave32), f16 intermediates.
// Kernel 2 uses double-buffered GLOBAL_LOAD_ASYNC_TO_LDS when available.
// ---------------------------------------------------------------------------

typedef __attribute__((ext_vector_type(16))) _Float16 v16h;
typedef __attribute__((ext_vector_type(8)))  _Float16 v8h;
typedef __attribute__((ext_vector_type(8)))  float    v8f;
typedef __attribute__((ext_vector_type(4)))  int      v4i;

constexpr int S_DIM = 512;
constexpr int ZD    = 128;                 // Z == E == 128
constexpr int MTOT  = S_DIM * S_DIM;       // 262144 rows (s,r) flattened

#if defined(__has_builtin)
#if __has_builtin(__builtin_amdgcn_global_load_async_to_lds_b128)
#define USE_ASYNC_LDS 1
#endif
#if __has_builtin(__builtin_amdgcn_s_wait_asynccnt)
#define HAVE_WAIT_ASYNC_BUILTIN 1
#endif
#endif

__device__ __forceinline__ void wait_async0() {
#ifdef HAVE_WAIT_ASYNC_BUILTIN
  __builtin_amdgcn_s_wait_asynccnt(0);
#else
  asm volatile("s_wait_asynccnt 0x0" ::: "memory");
#endif
}

union HF16 { v16h v; v8h h[2]; };

// A fragment, 16x32 f16. Lane L (L<16): row M=L, K = 0..7 & 16..23;
// lane L+16: row M=L, K = 8..15 & 24..31. Two 16B contiguous chunks per lane.
__device__ __forceinline__ v16h frag_a(const _Float16* base, int ld) {
  const int lane = threadIdx.x & 31;
  const int m  = lane & 15;
  const int hi = lane >> 4;
  const _Float16* p = base + (size_t)m * ld + hi * 8;
  HF16 f;
  f.h[0] = *(const v8h*)(p);
  f.h[1] = *(const v8h*)(p + 16);
  return f.v;
}

// B fragment, 32x16 f16, stored as B^T rows (row n = column n of B, K-contig).
__device__ __forceinline__ v16h frag_b(const _Float16* base, int ld) {
  const int lane = threadIdx.x & 31;
  const int n  = lane & 15;
  const int hi = lane >> 4;
  const _Float16* p = base + (size_t)n * ld + hi * 8;
  HF16 f;
  f.h[0] = *(const v8h*)(p);
  f.h[1] = *(const v8h*)(p + 16);
  return f.v;
}

__device__ __forceinline__ v8f wmma_f16(v16h a, v16h b, v8f c) {
  return __builtin_amdgcn_wmma_f32_16x16x32_f16(false, a, false, b,
                                                (short)0, c, false, false);
}

__device__ __forceinline__ v8f v8f_zero() {
  v8f z = {0.f, 0.f, 0.f, 0.f, 0.f, 0.f, 0.f, 0.f};
  return z;
}

// Transposed f16 store of a 16x16 C tile (lane = column, 8 consecutive rows).
__device__ __forceinline__ void store_ct(const v8f& acc, _Float16* dstT, size_t ld) {
  const int lane = threadIdx.x & 31;
  const int n  = lane & 15;
  const int mb = (lane >> 4) * 8;
  v8h o;
#pragma unroll
  for (int i = 0; i < 8; ++i) o[i] = (_Float16)acc[i];
  *(v8h*)(dstT + (size_t)n * ld + mb) = o;
}

__device__ __forceinline__ float sigm(float x) { return 1.0f / (1.0f + __expf(-x)); }

// Stage one 32-f16 (64B) row global -> LDS.
__device__ __forceinline__ void stage_row(const _Float16* g, _Float16* l) {
#ifdef USE_ASYNC_LDS
#pragma unroll
  for (int q = 0; q < 4; ++q)
    __builtin_amdgcn_global_load_async_to_lds_b128((v4i*)(g + q * 8),
                                                   (v4i*)(l + q * 8), 0, 0);
#else
#pragma unroll
  for (int q = 0; q < 4; ++q)
    *(v8h*)(l + q * 8) = *(const v8h*)(g + q * 8);
#endif
}

// ---------------------------------------------------------------------------
// Kernel 0: convert + transpose six 128x128 f32 weights to f16 B^T layout.
// ---------------------------------------------------------------------------
__global__ void k0_cvt_w(const float* __restrict__ l1, const float* __restrict__ l2,
                         const float* __restrict__ r1, const float* __restrict__ r2,
                         const float* __restrict__ g,  const float* __restrict__ o,
                         _Float16* __restrict__ wt) {
  const float* srcs[6] = {l1, l2, r1, r2, g, o};
  const float* s = srcs[blockIdx.x];
  _Float16* d = wt + (size_t)blockIdx.x * (ZD * ZD);
  for (int idx = threadIdx.x; idx < ZD * ZD; idx += blockDim.x) {
    const int k = idx >> 7;
    const int n = idx & 127;
    d[n * ZD + k] = (_Float16)s[idx];
  }
}

// ---------------------------------------------------------------------------
// Kernel 1: LN over Z + five fused 128->128 projections + activations.
// Outputs written TRANSPOSED as f16: leftT/rightT/gateT [e][m].
// ---------------------------------------------------------------------------
__global__ void __launch_bounds__(256)
k1_ln_proj(const float* __restrict__ X,
           const float* __restrict__ lnw, const float* __restrict__ lnb,
           const _Float16* __restrict__ wt5,
           _Float16* __restrict__ lTT, _Float16* __restrict__ rTT,
           _Float16* __restrict__ gTT) {
  __shared__ _Float16 xt[16][136];
  __shared__ float red[2][16][16];
  const int m0 = blockIdx.x * 16;
  const int t = threadIdx.x;
  const int row = t >> 4, part = t & 15;
  const float* xr = X + (size_t)(m0 + row) * ZD + part * 8;
  float v[8];
  float s = 0.f, s2 = 0.f;
#pragma unroll
  for (int i = 0; i < 8; ++i) { v[i] = xr[i]; s += v[i]; s2 += v[i] * v[i]; }
  red[0][row][part] = s; red[1][row][part] = s2;
  __syncthreads();
  float ts = 0.f, ts2 = 0.f;
#pragma unroll
  for (int i = 0; i < 16; ++i) { ts += red[0][row][i]; ts2 += red[1][row][i]; }
  const float mu = ts * (1.f / ZD);
  const float rs = rsqrtf(ts2 * (1.f / ZD) - mu * mu + 1e-5f);
#pragma unroll
  for (int i = 0; i < 8; ++i) {
    const int z = part * 8 + i;
    xt[row][z] = (_Float16)((v[i] - mu) * rs * lnw[z] + lnb[z]);
  }
  __syncthreads();

  const int wave = t >> 5;
  const int n0 = wave * 16;
  v8f acc[5];
#pragma unroll
  for (int mm = 0; mm < 5; ++mm) acc[mm] = v8f_zero();
#pragma unroll
  for (int k0 = 0; k0 < ZD; k0 += 32) {
    const v16h af = frag_a(&xt[0][0] + k0, 136);
    v16h bfs[5];                     // batch B loads -> one clause, then 5 WMMAs
#pragma unroll
    for (int mm = 0; mm < 5; ++mm)
      bfs[mm] = frag_b(wt5 + (size_t)mm * (ZD * ZD) + n0 * ZD + k0, ZD);
#pragma unroll
    for (int mm = 0; mm < 5; ++mm) acc[mm] = wmma_f16(af, bfs[mm], acc[mm]);
  }
  v8f lv, rv, gv;
#pragma unroll
  for (int i = 0; i < 8; ++i) {
    lv[i] = sigm(acc[0][i]) * acc[1][i];
    rv[i] = sigm(acc[2][i]) * acc[3][i];
    gv[i] = sigm(acc[4][i]);
  }
  store_ct(lv, lTT + (size_t)n0 * MTOT + m0, MTOT);
  store_ct(rv, rTT + (size_t)n0 * MTOT + m0, MTOT);
  store_ct(gv, gTT + (size_t)n0 * MTOT + m0, MTOT);
}

// ---------------------------------------------------------------------------
// Kernel 2: triangle einsum as 128 batched GEMMs P_e = L_e * R_e^T.
// Double-buffered LDS, async global->LDS DMA overlapped with WMMA compute.
// Output stored pT[e][k][s] = p[s,k,e] via transposed C stores.
// ---------------------------------------------------------------------------
__global__ void __launch_bounds__(256)
k2_triangle(const _Float16* __restrict__ lT,
            const _Float16* __restrict__ rT,
            _Float16* __restrict__ pT) {
  __shared__ _Float16 At[2][128][40];
  __shared__ _Float16 Bt[2][128][40];
  const int e   = blockIdx.z;
  const int s0  = blockIdx.x * 128;
  const int k0b = blockIdx.y * 128;
  const size_t eoff = (size_t)e * MTOT;
  const int t = threadIdx.x;
  const int wave = t >> 5;
  const int wy = wave >> 2;                 // 0..1 -> M offset wy*64
  const int wx = wave & 3;                  // 0..3 -> N offset wx*32

  const int  crow = t & 127;
  const bool isB  = t >= 128;
  const _Float16* gsrc = isB ? rT + eoff + (size_t)(k0b + crow) * S_DIM
                             : lT + eoff + (size_t)(s0  + crow) * S_DIM;
  _Float16* ldst0 = isB ? &Bt[0][crow][0] : &At[0][crow][0];
  _Float16* ldst1 = isB ? &Bt[1][crow][0] : &At[1][crow][0];

  v8f acc[4][2];
#pragma unroll
  for (int i = 0; i < 4; ++i)
#pragma unroll
    for (int j = 0; j < 2; ++j) acc[i][j] = v8f_zero();

  constexpr int NST = S_DIM / 32;           // 16 K-slabs
  stage_row(gsrc, ldst0);                   // prologue: stage 0
  for (int st = 0; st < NST; ++st) {
    const int c = st & 1;
#ifdef USE_ASYNC_LDS
    wait_async0();                          // own stage-st DMAs complete
#endif
    __syncthreads();                        // whole tile visible; prev reads done
    if (st + 1 < NST)
      stage_row(gsrc + (st + 1) * 32, c ? ldst0 : ldst1);  // overlap with compute

    v16h bf[2];
#pragma unroll
    for (int j = 0; j < 2; ++j) bf[j] = frag_b(&Bt[c][wx * 32 + j * 16][0], 40);
#pragma unroll
    for (int i = 0; i < 4; ++i) {
      const v16h af = frag_a(&At[c][wy * 64 + i * 16][0], 40);
#pragma unroll
      for (int j = 0; j < 2; ++j) acc[i][j] = wmma_f16(af, bf[j], acc[i][j]);
    }
  }

#pragma unroll
  for (int i = 0; i < 4; ++i)
#pragma unroll
    for (int j = 0; j < 2; ++j)
      store_ct(acc[i][j],
               pT + eoff + (size_t)(k0b + wx * 32 + j * 16) * S_DIM
                  + (s0 + wy * 64 + i * 16),
               S_DIM);
}

// ---------------------------------------------------------------------------
// Kernel 3: LN2 over e + output GEMM (E->Z) + gate multiply + residual.
// A fragments hoisted (8 frags/wave); B fragments loaded once per N-tile.
// ---------------------------------------------------------------------------
__global__ void __launch_bounds__(256)
k3_out(const _Float16* __restrict__ pT,
       const float* __restrict__ ln2w, const float* __restrict__ ln2b,
       const _Float16* __restrict__ wtO, const _Float16* __restrict__ gTT,
       const float* __restrict__ X, float* __restrict__ out) {
  __shared__ _Float16 pln[256][136];
  const int s0 = blockIdx.x * 16, r0 = blockIdx.y * 16;
  const int t = threadIdx.x;
  const int j = t & 15, i = t >> 4;          // i = s_local, j = r_local
  const int row = i * 16 + j;
  const size_t base = (size_t)(s0 + i) * S_DIM + (r0 + j);
  float s = 0.f, s2 = 0.f;
  for (int e = 0; e < ZD; ++e) {
    if (e + 1 < ZD) __builtin_prefetch(pT + (size_t)(e + 1) * MTOT + base, 0, 1);
    const float v = (float)pT[(size_t)e * MTOT + base];
    pln[row][e] = (_Float16)v;
    s += v; s2 += v * v;
  }
  const float mu = s * (1.f / ZD);
  const float rs = rsqrtf(s2 * (1.f / ZD) - mu * mu + 1e-5f);
  for (int e = 0; e < ZD; ++e) {
    const float v = (float)pln[row][e];
    pln[row][e] = (_Float16)((v - mu) * rs * ln2w[e] + ln2b[e]);
  }
  __syncthreads();

  const int wave = t >> 5;
  const int lane = t & 31;
  const int cn = lane & 15, mb = (lane >> 4) * 8;

  v16h afr[2][4];                             // hoisted A fragments
#pragma unroll
  for (int mi = 0; mi < 2; ++mi)
#pragma unroll
    for (int k = 0; k < 4; ++k)
      afr[mi][k] = frag_a(&pln[(wave * 2 + mi) * 16][0] + k * 32, 136);

  for (int nt = 0; nt < 8; ++nt) {
    v16h bf[4];
#pragma unroll
    for (int k = 0; k < 4; ++k) bf[k] = frag_b(wtO + nt * 16 * ZD + k * 32, ZD);
#pragma unroll
    for (int mi = 0; mi < 2; ++mi) {
      const int mt = wave * 2 + mi;           // M-tile: s_local = mt, rows = r
      v8f acc = v8f_zero();
#pragma unroll
      for (int k = 0; k < 4; ++k) acc = wmma_f16(afr[mi][k], bf[k], acc);

      const int z = nt * 16 + cn;
      const size_t mg = (size_t)(s0 + mt) * S_DIM + r0 + mb; // 8 consecutive r
      const v8h gvec = *(const v8h*)(gTT + (size_t)z * MTOT + mg);
#pragma unroll
      for (int vv = 0; vv < 8; ++vv) {
        const size_t oi = (mg + vv) * ZD + z;
        out[oi] = X[oi] + acc[vv] * (float)gvec[vv];
      }
    }
  }
}

// ---------------------------------------------------------------------------
extern "C" void kernel_launch(void* const* d_in, const int* in_sizes, int n_in,
                              void* d_out, int out_size, void* d_ws, size_t ws_size,
                              hipStream_t stream) {
  (void)in_sizes; (void)n_in; (void)out_size; (void)ws_size;
  const float* pair = (const float*)d_in[0];
  const float* ln1w = (const float*)d_in[1];
  const float* ln1b = (const float*)d_in[2];
  const float* l1   = (const float*)d_in[3];
  const float* l2   = (const float*)d_in[4];
  const float* r1   = (const float*)d_in[5];
  const float* r2   = (const float*)d_in[6];
  const float* ln2w = (const float*)d_in[7];
  const float* ln2b = (const float*)d_in[8];
  const float* ow   = (const float*)d_in[9];
  const float* gw   = (const float*)d_in[10];
  float* out = (float*)d_out;

  char* ws = (char*)d_ws;
  _Float16* wt = (_Float16*)ws;                        // 6 * 128*128 f16
  _Float16* gT = (_Float16*)(ws + (size_t)(1u << 20)); // [128][262144] f16
  _Float16* lT = gT + (size_t)ZD * MTOT;
  _Float16* rT = lT + (size_t)ZD * MTOT;
  _Float16* pT = rT + (size_t)ZD * MTOT;

  k0_cvt_w<<<6, 256, 0, stream>>>(l1, l2, r1, r2, gw, ow, wt);
  k1_ln_proj<<<MTOT / 16, 256, 0, stream>>>(pair, ln1w, ln1b, wt, lT, rT, gT);
  dim3 g2(4, 4, 128);
  k2_triangle<<<g2, 256, 0, stream>>>(lT, rT, pT);
  dim3 g3(S_DIM / 16, S_DIM / 16);
  k3_out<<<g3, 256, 0, stream>>>(pT, ln2w, ln2b, wt + 5 * (ZD * ZD), gT, pair, out);
}
